// Model_8263517078167
// MI455X (gfx1250) — matrix operand order repaired and checked by reading, not compiled, not run on hardware
//
#include <hip/hip_runtime.h>

// NeRF density: 16-level hash-grid encode (F=2) + MLP 32->64(ReLU)->1 + exp, masked.
// Phase 1: 1 thread/point gathers from 64MB tables (L2-resident on MI455X) -> f16 enc in LDS.
// Phase 2: per-wave WMMA f32_16x16x32_f16 for the 32->64 layer; layer 2 folded into epilogue.

typedef __attribute__((ext_vector_type(16))) _Float16 v16h;
typedef __attribute__((ext_vector_type(8)))  _Float16 v8h;
typedef __attribute__((ext_vector_type(8)))  float    v8f;

#define NLVL   16
#define LOG2T  19
#define TSIZE  (1u << LOG2T)
#define ROWS   40   // padded LDS row stride in halfs (80B, 16B-aligned chunks)

__global__ __launch_bounds__(256)
void nerf_density_fused(const float* __restrict__ pos,
                        const float* __restrict__ tables,
                        const float* __restrict__ W1,
                        const float* __restrict__ b1,
                        const float* __restrict__ W2,
                        const float* __restrict__ b2,
                        float* __restrict__ out,
                        int n)
{
    __shared__ _Float16 encS[256 * ROWS];
    __shared__ float    selS[256];

    const int tid = threadIdx.x;
    const int gi  = blockIdx.x * 256 + tid;

    // ---------------- Phase 1: hash-grid encoding ----------------
    float px = 0.f, py = 0.f, pz = 0.f, sel = 0.f;
    if (gi < n) {
        float x = pos[3*gi+0], y = pos[3*gi+1], z = pos[3*gi+2];
        px = (x + 2.0f) * 0.25f;
        py = (y + 2.0f) * 0.25f;
        pz = (z + 2.0f) * 0.25f;
        bool inb = (px >= 0.f) & (px <= 1.f) & (py >= 0.f) & (py <= 1.f)
                 & (pz >= 0.f) & (pz <= 1.f);
        sel = inb ? 1.f : 0.f;
        px *= sel; py *= sel; pz *= sel;
    }
    selS[tid] = sel;

    #pragma unroll 1
    for (int l = 0; l < NLVL; ++l) {
        // scale = 16 * 128^(l/15) - 1 = 2^(4 + 7l/15) - 1   (exact)
        float scale = exp2f(4.0f + (7.0f / 15.0f) * (float)l) - 1.0f;
        int   res   = (int)ceilf(scale) + 1;
        bool  dense = ((long long)res * res * res) <= (long long)TSIZE;

        float xsx = px * scale + 0.5f;
        float xsy = py * scale + 0.5f;
        float xsz = pz * scale + 0.5f;
        float fix = floorf(xsx), fiy = floorf(xsy), fiz = floorf(xsz);
        float fx = xsx - fix, fy = xsy - fiy, fz = xsz - fiz;
        int ix = (int)fix, iy = (int)fiy, iz = (int)fiz;

        float f0 = 0.f, f1 = 0.f;
        const float2* tb = (const float2*)tables + ((size_t)l << LOG2T);
        #pragma unroll
        for (int c = 0; c < 8; ++c) {
            int ox = (c >> 2) & 1, oy = (c >> 1) & 1, oz = c & 1;
            unsigned idx;
            if (dense) {
                int cx = ix + ox; cx = cx < 0 ? 0 : (cx > res-1 ? res-1 : cx);
                int cy = iy + oy; cy = cy < 0 ? 0 : (cy > res-1 ? res-1 : cy);
                int cz = iz + oz; cz = cz < 0 ? 0 : (cz > res-1 ? res-1 : cz);
                idx = (unsigned)(cx + res * (cy + res * cz));
            } else {
                unsigned ux = (unsigned)(ix + ox);
                unsigned uy = (unsigned)(iy + oy);
                unsigned uz = (unsigned)(iz + oz);
                unsigned h = ux ^ (uy * 2654435761u) ^ (uz * 805459861u);
                idx = h & (TSIZE - 1u);
            }
            float w = (ox ? fx : 1.f - fx) * (oy ? fy : 1.f - fy) * (oz ? fz : 1.f - fz);
            float2 t = tb[idx];
            f0 = fmaf(w, t.x, f0);
            f1 = fmaf(w, t.y, f1);
        }
        encS[tid * ROWS + 2*l]     = (_Float16)f0;
        encS[tid * ROWS + 2*l + 1] = (_Float16)f1;
    }
    __syncthreads();

    // ---------------- Phase 2: MLP via WMMA ----------------
    const int lane    = tid & 31;
    const int wave    = tid >> 5;
    const int halfSel = lane >> 4;    // 0: lanes 0-15, 1: lanes 16-31
    const int l15     = lane & 15;

    // B tiles: B[k][nc] = W1[nc][k] (W1 is 64x32 row-major).
    // 16-bit B 32x16 layout: lane -> column; lanes 0-15 hold K=0..7 then 16..23, lanes 16-31 hold K=8..15 then 24..31.
    v16h Bt[4];
    float b1v[4], w2v[4];
    #pragma unroll
    for (int nt = 0; nt < 4; ++nt) {
        int nc = nt * 16 + l15;
        const float* wr = W1 + nc * 32 + halfSel * 8;
        v16h bb;
        #pragma unroll
        for (int i = 0; i < 16; ++i) bb[i] = (_Float16)wr[i < 8 ? i : i + 8];
        Bt[nt]  = bb;
        b1v[nt] = b1[nc];
        w2v[nt] = W2[nc];
    }
    const float b2v = b2[0];

    float red[2][8];
    #pragma unroll
    for (int gg = 0; gg < 2; ++gg) {
        const int grp  = wave * 2 + gg;
        const int mrow = grp * 16 + l15;
        // A 16x32 f16 layout: lanes 0-15: K=0..7 then 16..23; lanes 16-31: K=8..15 then 24..31.
        const _Float16* ar = &encS[mrow * ROWS + halfSel * 8];
        v8h alo = *(const v8h*)(ar);
        v8h ahi = *(const v8h*)(ar + 16);
        v16h A;
        #pragma unroll
        for (int i = 0; i < 8; ++i) { A[i] = alo[i]; A[8+i] = ahi[i]; }

        float acc[8] = {0,0,0,0,0,0,0,0};
        #pragma unroll
        for (int nt = 0; nt < 4; ++nt) {
            v8f c = {0,0,0,0,0,0,0,0};
            c = __builtin_amdgcn_wmma_f32_16x16x32_f16(false, A, false, Bt[nt],
                                                       (short)0, c, false, false);
            // c[r]: row M = r + 8*halfSel, col N = nt*16 + l15
            #pragma unroll
            for (int r = 0; r < 8; ++r) {
                float h = c[r] + b1v[nt];
                h = h > 0.f ? h : 0.f;
                acc[r] = fmaf(h, w2v[nt], acc[r]);   // fold layer-2 dot product in
            }
        }
        // reduce across the 16 lanes of each half (xor<16 never crosses halves)
        #pragma unroll
        for (int r = 0; r < 8; ++r) {
            float v = acc[r];
            v += __shfl_xor(v, 1);
            v += __shfl_xor(v, 2);
            v += __shfl_xor(v, 4);
            v += __shfl_xor(v, 8);
            red[gg][r] = v;
        }
    }

    #pragma unroll
    for (int gg = 0; gg < 2; ++gg) {
        if (l15 == 0) {
            const int mb = (wave * 2 + gg) * 16 + halfSel * 8;
            #pragma unroll
            for (int r = 0; r < 8; ++r) {
                int m = mb + r;
                int g = blockIdx.x * 256 + m;
                if (g < n) out[g] = selS[m] * expf(red[gg][r] + b2v);
            }
        }
    }
}

extern "C" void kernel_launch(void* const* d_in, const int* in_sizes, int n_in,
                              void* d_out, int out_size, void* d_ws, size_t ws_size,
                              hipStream_t stream) {
    const float* pos    = (const float*)d_in[0];
    // d_in[1] = viewdirs (unused by the reference output)
    const float* tables = (const float*)d_in[2];
    const float* W1     = (const float*)d_in[3];
    const float* b1     = (const float*)d_in[4];
    const float* W2     = (const float*)d_in[5];
    const float* b2     = (const float*)d_in[6];
    float* out = (float*)d_out;
    (void)d_ws; (void)ws_size; (void)n_in;

    const int n = in_sizes[0] / 3;
    const int nblocks = (n + 255) / 256;
    nerf_density_fused<<<nblocks, 256, 0, stream>>>(pos, tables, W1, b1, W2, b2, out, n);
}
